// Indexer_71657234367033
// MI455X (gfx1250) — compile-verified, matrix operand order checked by reading, not run on hardware
//
#include <hip/hip_runtime.h>
#include <hip/hip_bf16.h>

// ---------------- types / helpers ----------------
typedef __attribute__((ext_vector_type(16))) __bf16 v16bf;
typedef __attribute__((ext_vector_type(8)))  float  v8f;
typedef __attribute__((ext_vector_type(4)))  unsigned int u32x4;

union Frag16 { u32x4 q[2]; v16bf v; };

__device__ __forceinline__ unsigned short f2bf(float f) {
    unsigned u = __float_as_uint(f);
    unsigned r = u + 0x7FFFu + ((u >> 16) & 1u);   // round-to-nearest-even
    return (unsigned short)(r >> 16);
}
__device__ __forceinline__ float bf2f(unsigned short h) {
    return __uint_as_float(((unsigned)h) << 16);
}
// order-preserving float<->uint maps for sorting
__device__ __forceinline__ unsigned mapf(float f) {
    unsigned u = __float_as_uint(f);
    return (u & 0x80000000u) ? ~u : (u | 0x80000000u);
}
__device__ __forceinline__ float unmapf(unsigned m) {
    unsigned u = (m & 0x80000000u) ? (m & 0x7FFFFFFFu) : ~m;
    return __uint_as_float(u);
}

#define S_  2048
#define D_  7168
#define H_  64
#define HD_ 128
#define QD_ 8192          // H_*HD_
#define QLR_ 1536
#define KW_N 192          // HD_ + H_ fused output width
#define ROPE_HALF 32
#define TOPK_ 512
#define NEG_INF_ (-1e30f)
#define WSCALE_ 0.011048543456039804f   // 64^-0.5 * 128^-0.5

// ---------------- 1a. generic fp32 -> bf16 ----------------
__global__ void __launch_bounds__(256) cvt_bf16_kernel(const float* __restrict__ in,
                                                       unsigned short* __restrict__ out, int n) {
    int i = blockIdx.x * 256 + threadIdx.x;
    if (i < n) out[i] = f2bf(in[i]);
}

// ---------------- 1b. generic tiled transpose + fp32->bf16 ----------------
// out[(outRowBase + c)*outStride + r] = bf16(in[r*inCols + c])
// grid = (inCols/32, inRows/32), block = (32, 8); dims must be multiples of 32.
__global__ void __launch_bounds__(256) transpose_cvt_kernel(const float* __restrict__ in,
                                                            unsigned short* __restrict__ out,
                                                            int inCols, int outStride,
                                                            int outRowBase) {
    __shared__ float tile[32][33];
    int cBase = blockIdx.x * 32;
    int rBase = blockIdx.y * 32;
    int tx = threadIdx.x, ty = threadIdx.y;
    #pragma unroll
    for (int yy = 0; yy < 4; ++yy) {
        int r = ty * 4 + yy;
        tile[r][tx] = in[(size_t)(rBase + r) * inCols + (cBase + tx)];
    }
    __syncthreads();
    #pragma unroll
    for (int yy = 0; yy < 4; ++yy) {
        int r = ty * 4 + yy;
        out[(size_t)(outRowBase + cBase + r) * outStride + (rBase + tx)] = f2bf(tile[tx][r]);
    }
}

// ---------------- 2. fused k/w GEMM: [k | w] = x @ [wk | wp], WMMA bf16 ------------
// M=2048, N=192, K=7168. Block = 4 waves, each wave computes 16x48 (3 accumulators).
__global__ void __launch_bounds__(128) kw_gemm_kernel(const unsigned short* __restrict__ xbf,
                                                      const unsigned short* __restrict__ kwT,
                                                      float* __restrict__ kwRaw) {
    int sBase = blockIdx.x * 16;
    int tid = threadIdx.x;
    int wave = tid >> 5, lane = tid & 31, l = lane & 15, hi = lane >> 4;
    int nBase = wave * 48;
    v8f zero = {0,0,0,0,0,0,0,0};
    v8f acc[3] = {zero, zero, zero};
    const unsigned short* arow = xbf + (size_t)(sBase + l) * D_;
    for (int k0 = 0; k0 < D_; k0 += 32) {
        Frag16 af;
        af.q[0] = *(const u32x4*)(arow + k0 + hi * 8);
        af.q[1] = *(const u32x4*)(arow + k0 + 16 + hi * 8);
        #pragma unroll
        for (int c = 0; c < 3; ++c) {
            Frag16 bf;
            const unsigned short* brow = kwT + (size_t)(nBase + c * 16 + l) * D_;
            bf.q[0] = *(const u32x4*)(brow + k0 + hi * 8);
            bf.q[1] = *(const u32x4*)(brow + k0 + 16 + hi * 8);
            acc[c] = __builtin_amdgcn_wmma_f32_16x16x32_bf16(
                false, af.v, false, bf.v, (short)0, acc[c], false, false);
        }
    }
    #pragma unroll
    for (int c = 0; c < 3; ++c)
        #pragma unroll
        for (int r = 0; r < 8; ++r) {
            int row = sBase + (hi ? r + 8 : r);
            int col = nBase + c * 16 + l;
            kwRaw[(size_t)row * KW_N + col] = acc[c][r];
        }
}

// ---------------- 3. epilogue: LN + rope on k (->bf16), scale w (fp32) -------------
// One wave per row; lane owns dims 4l..4l+3 (rope pairs are lane-local).
__global__ void __launch_bounds__(256) ln_rope_w_kernel(const float* __restrict__ kwRaw,
                                                        const float* __restrict__ knw,
                                                        const float* __restrict__ knb,
                                                        const float* __restrict__ cosT,
                                                        const float* __restrict__ sinT,
                                                        unsigned short* __restrict__ kbf,
                                                        float* __restrict__ wF) {
    int wave = threadIdx.x >> 5, lane = threadIdx.x & 31;
    int row = blockIdx.x * 8 + wave;
    const float* src = kwRaw + (size_t)row * KW_N;
    float v[4];
    #pragma unroll
    for (int j = 0; j < 4; ++j) v[j] = src[lane * 4 + j];
    float sum = v[0] + v[1] + v[2] + v[3];
    #pragma unroll
    for (int off = 16; off > 0; off >>= 1) sum += __shfl_xor(sum, off, 32);
    float mu = sum * (1.f / HD_);
    float sq = 0.f;
    #pragma unroll
    for (int j = 0; j < 4; ++j) { float d = v[j] - mu; sq += d * d; }
    #pragma unroll
    for (int off = 16; off > 0; off >>= 1) sq += __shfl_xor(sq, off, 32);
    float rs = rsqrtf(sq * (1.f / HD_) + 1e-6f);
    float n[4];
    #pragma unroll
    for (int j = 0; j < 4; ++j) {
        int dim = lane * 4 + j;
        n[j] = (v[j] - mu) * rs * knw[dim] + knb[dim];
    }
    unsigned short o[4];
    if (lane < 16) {   // dims < 64: interleaved rope; pairs (4l,4l+1),(4l+2,4l+3)
        int i0 = lane * 2, i1 = lane * 2 + 1;
        float c0 = cosT[row * ROPE_HALF + i0], s0 = sinT[row * ROPE_HALF + i0];
        float c1 = cosT[row * ROPE_HALF + i1], s1 = sinT[row * ROPE_HALF + i1];
        o[0] = f2bf(n[0] * c0 - n[1] * s0);
        o[1] = f2bf(n[0] * s0 + n[1] * c0);
        o[2] = f2bf(n[2] * c1 - n[3] * s1);
        o[3] = f2bf(n[2] * s1 + n[3] * c1);
    } else {
        #pragma unroll
        for (int j = 0; j < 4; ++j) o[j] = f2bf(n[j]);
    }
    #pragma unroll
    for (int j = 0; j < 4; ++j) kbf[(size_t)row * HD_ + lane * 4 + j] = o[j];
    // head weights: cols 128..191, two per lane
    wF[(size_t)row * H_ + lane]      = src[HD_ + lane]      * WSCALE_;
    wF[(size_t)row * H_ + 32 + lane] = src[HD_ + 32 + lane] * WSCALE_;
}

// ---------------- 4. q = q_lora @ wq_b : WMMA bf16, wave = 16x64 tile ----------------
__global__ void __launch_bounds__(128) gemm_q_kernel(const unsigned short* __restrict__ ql,
                                                     const unsigned short* __restrict__ wqT,
                                                     unsigned short* __restrict__ qout) {
    int sBase = blockIdx.y * 16;
    int tid = threadIdx.x;
    int wave = tid >> 5, lane = tid & 31, l = lane & 15, hi = lane >> 4;
    __shared__ unsigned short aT[16 * QLR_];                  // 48 KB A tile
    {
        const u32x4* src = (const u32x4*)(ql + (size_t)sBase * QLR_);
        u32x4* dst = (u32x4*)aT;
        for (int i = tid; i < 16 * QLR_ / 8; i += 128) dst[i] = src[i];
    }
    __syncthreads();
    int nBase = blockIdx.x * 256 + wave * 64;
    v8f zero = {0,0,0,0,0,0,0,0};
    v8f acc[4] = {zero, zero, zero, zero};
    for (int k0 = 0; k0 < QLR_; k0 += 32) {
        Frag16 af;
        af.q[0] = *(const u32x4*)&aT[l * QLR_ + k0 + hi * 8];
        af.q[1] = *(const u32x4*)&aT[l * QLR_ + k0 + 16 + hi * 8];
        #pragma unroll
        for (int c = 0; c < 4; ++c) {
            Frag16 bf;
            const unsigned short* brow = wqT + (size_t)(nBase + c * 16 + l) * QLR_;
            bf.q[0] = *(const u32x4*)(brow + k0 + hi * 8);
            bf.q[1] = *(const u32x4*)(brow + k0 + 16 + hi * 8);
            acc[c] = __builtin_amdgcn_wmma_f32_16x16x32_bf16(
                false, af.v, false, bf.v, (short)0, acc[c], false, false);
        }
    }
    #pragma unroll
    for (int c = 0; c < 4; ++c)
        #pragma unroll
        for (int r = 0; r < 8; ++r) {
            int row = sBase + (hi ? r + 8 : r);
            int col = nBase + c * 16 + l;
            qout[(size_t)row * QD_ + col] = f2bf(acc[c][r]);
        }
}

// ---------------- 5. interleaved rope on q (first 64 dims per head) ----------------
__global__ void __launch_bounds__(256) rope_q_kernel(unsigned short* __restrict__ qbf,
                                                     const float* __restrict__ cosT,
                                                     const float* __restrict__ sinT) {
    int gid = blockIdx.x * 256 + threadIdx.x;
    const int total = S_ * H_ * ROPE_HALF;
    if (gid >= total) return;
    int i = gid & 31, h = (gid >> 5) & 63, s = gid >> 11;
    size_t base = (size_t)s * QD_ + h * HD_ + 2 * i;
    float x1 = bf2f(qbf[base]), x2 = bf2f(qbf[base + 1]);
    float c = cosT[s * ROPE_HALF + i], sn = sinT[s * ROPE_HALF + i];
    qbf[base]     = f2bf(x1 * c - x2 * sn);
    qbf[base + 1] = f2bf(x1 * sn + x2 * c);
}

// ---------------- 6a. fill score with NEG_INF ----------------
__global__ void __launch_bounds__(256) init_score_kernel(float* __restrict__ score) {
    int i = blockIdx.x * 256 + threadIdx.x;
    if (i < S_ * S_) score[i] = NEG_INF_;
}

// ---------------- 6b. fused scores: sum_h w[s,h]*relu(q[s,h,:]·k[t,:]) + causal ----------
// block = 16-row s-tile, 8 waves; each wave owns one 16x16 output tile per iteration.
// K-tile B fragments live in 32 VGPRs reused across all 64 heads (4 WMMAs/head, K=128).
__global__ void __launch_bounds__(256) scores_kernel(const unsigned short* __restrict__ qbf,
                                                     const unsigned short* __restrict__ kbf,
                                                     const float* __restrict__ wF,
                                                     float* __restrict__ score) {
    int sTile = blockIdx.x;
    int sBase = sTile * 16;
    int tid = threadIdx.x;
    int wave = tid >> 5, lane = tid & 31, l = lane & 15, hi = lane >> 4;
    __shared__ float wS[16 * H_];                  // 4 KB head-weight tile
    for (int i = tid; i < 16 * H_; i += 256) {
        int r = i >> 6, h = i & 63;
        wS[i] = wF[(sBase + r) * H_ + h];
    }
    __syncthreads();
    int nT = sTile + 1;                            // causal: only t-tiles <= s-tile
    for (int tt = wave; tt < nT; tt += 8) {
        int tBase = tt * 16;
        Frag16 bfr[4];
        const unsigned short* krow = kbf + (size_t)(tBase + l) * HD_;
        #pragma unroll
        for (int kf = 0; kf < 4; ++kf) {
            int k0 = kf * 32;
            bfr[kf].q[0] = *(const u32x4*)(krow + k0 + hi * 8);
            bfr[kf].q[1] = *(const u32x4*)(krow + k0 + 16 + hi * 8);
        }
        v8f out = {0,0,0,0,0,0,0,0};
        const unsigned short* arowBase = qbf + (size_t)(sBase + l) * QD_;
        for (int h = 0; h < H_; ++h) {
            const unsigned short* arow = arowBase + h * HD_;
            __builtin_prefetch(arow + HD_, 0, 3);  // next head's A row
            v8f acc = {0,0,0,0,0,0,0,0};
            #pragma unroll
            for (int kf = 0; kf < 4; ++kf) {
                Frag16 af;
                int k0 = kf * 32;
                af.q[0] = *(const u32x4*)(arow + k0 + hi * 8);
                af.q[1] = *(const u32x4*)(arow + k0 + 16 + hi * 8);
                acc = __builtin_amdgcn_wmma_f32_16x16x32_bf16(
                    false, af.v, false, bfr[kf].v, (short)0, acc, false, false);
            }
            #pragma unroll
            for (int r = 0; r < 8; ++r) {
                int rowInTile = hi ? (r + 8) : r;
                float a = acc[r];
                out[r] += (a > 0.f ? a : 0.f) * wS[rowInTile * H_ + h];
            }
        }
        #pragma unroll
        for (int r = 0; r < 8; ++r) {
            int srow = sBase + (hi ? (r + 8) : r);
            int tcol = tBase + l;
            score[(size_t)srow * S_ + tcol] = (tcol <= srow) ? out[r] : NEG_INF_;
        }
    }
}

// ---------------- 7. per-row top-512 via bitonic sort (desc val, asc idx tiebreak) ---
__global__ void __launch_bounds__(256) topk_kernel(const float* __restrict__ score,
                                                   float* __restrict__ vals,
                                                   int* __restrict__ idx) {
    int row = blockIdx.x, tid = threadIdx.x;
    __shared__ unsigned long long keys[S_];        // 16 KB
    for (int i = tid; i < S_; i += 256) {
        unsigned m = mapf(score[(size_t)row * S_ + i]);
        keys[i] = ((unsigned long long)m << 32) | (unsigned)(~i);
    }
    __syncthreads();
    for (unsigned k = 2; k <= S_; k <<= 1) {
        for (unsigned j = k >> 1; j > 0; j >>= 1) {
            for (unsigned i = tid; i < S_; i += 256) {
                unsigned ixj = i ^ j;
                if (ixj > i) {
                    bool up = ((i & k) == 0);      // descending overall
                    unsigned long long a = keys[i], b = keys[ixj];
                    bool sw = up ? (a < b) : (a > b);
                    if (sw) { keys[i] = b; keys[ixj] = a; }
                }
            }
            __syncthreads();
        }
    }
    for (int i = tid; i < TOPK_; i += 256) {
        unsigned long long key = keys[i];
        vals[(size_t)row * TOPK_ + i] = unmapf((unsigned)(key >> 32));
        idx[(size_t)row * TOPK_ + i]  = (int)(~(unsigned)key);
    }
}

// ---------------- launch ----------------
extern "C" void kernel_launch(void* const* d_in, const int* in_sizes, int n_in,
                              void* d_out, int out_size, void* d_ws, size_t ws_size,
                              hipStream_t stream) {
    const float* x    = (const float*)d_in[0];   // (1,2048,7168)
    const float* qlo  = (const float*)d_in[1];   // (1,2048,1536)
    const float* wqb  = (const float*)d_in[2];   // (1536,8192)
    const float* wk   = (const float*)d_in[3];   // (7168,128)
    const float* wp   = (const float*)d_in[4];   // (7168,64)
    const float* knw  = (const float*)d_in[5];   // (128,)
    const float* knb  = (const float*)d_in[6];   // (128,)
    const float* cosT = (const float*)d_in[7];   // (2048,32)
    const float* sinT = (const float*)d_in[8];   // (2048,32)

    char* ws = (char*)d_ws;
    size_t off = 0;
    auto alloc = [&](size_t bytes) -> void* {
        void* p = ws + off;
        off = (off + bytes + 255) & ~(size_t)255;
        return p;
    };
    unsigned short* qloraBF = (unsigned short*)alloc((size_t)S_ * QLR_ * 2);
    unsigned short* wqT     = (unsigned short*)alloc((size_t)QD_ * QLR_ * 2);
    unsigned short* qBF     = (unsigned short*)alloc((size_t)S_ * QD_ * 2);
    unsigned short* xBF     = (unsigned short*)alloc((size_t)S_ * D_ * 2);
    unsigned short* kwT     = (unsigned short*)alloc((size_t)KW_N * D_ * 2);
    float*          kwRaw   = (float*)alloc((size_t)S_ * KW_N * 4);
    unsigned short* kBF     = (unsigned short*)alloc((size_t)S_ * HD_ * 2);
    float*          wF      = (float*)alloc((size_t)S_ * H_ * 4);
    float*          scoreF  = (float*)alloc((size_t)S_ * S_ * 4);

    float* vals = (float*)d_out;
    int*   idx  = (int*)d_out + (size_t)S_ * TOPK_;

    {   // 1a. fp32 -> bf16 conversions
        int n = S_ * QLR_;
        cvt_bf16_kernel<<<(n + 255) / 256, 256, 0, stream>>>(qlo, qloraBF, n);
        n = S_ * D_;
        cvt_bf16_kernel<<<(n + 255) / 256, 256, 0, stream>>>(x, xBF, n);
    }
    {   // 1b. transposed bf16 weights
        dim3 b(32, 8);
        transpose_cvt_kernel<<<dim3(QD_ / 32, QLR_ / 32), b, 0, stream>>>(wqb, wqT, QD_, QLR_, 0);
        transpose_cvt_kernel<<<dim3(HD_ / 32, D_ / 32),  b, 0, stream>>>(wk,  kwT, HD_, D_, 0);
        transpose_cvt_kernel<<<dim3(H_ / 32, D_ / 32),   b, 0, stream>>>(wp,  kwT, H_,  D_, HD_);
    }
    // 2. fused [k|w] GEMM (WMMA bf16)
    kw_gemm_kernel<<<S_ / 16, 128, 0, stream>>>(xBF, kwT, kwRaw);
    // 3. LN + rope + w-scale epilogue (wave per row)
    ln_rope_w_kernel<<<S_ / 8, 256, 0, stream>>>(kwRaw, knw, knb, cosT, sinT, kBF, wF);
    {   // 4. q GEMM (WMMA bf16)
        dim3 g(QD_ / 256, S_ / 16);
        gemm_q_kernel<<<g, 128, 0, stream>>>(qloraBF, wqT, qBF);
    }
    {   // 5. rope on q
        int n = S_ * H_ * ROPE_HALF;
        rope_q_kernel<<<(n + 255) / 256, 256, 0, stream>>>(qBF, cosT, sinT);
    }
    {   // 6a. NEG_INF fill
        int n = S_ * S_;
        init_score_kernel<<<(n + 255) / 256, 256, 0, stream>>>(scoreF);
    }
    // 6b. fused scores (WMMA bf16)
    scores_kernel<<<S_ / 16, 256, 0, stream>>>(qBF, kBF, wF, scoreF);
    // 7. top-k
    topk_kernel<<<S_, 256, 0, stream>>>(scoreF, vals, idx);
}
